// BNN_FMNIST_17609365914255
// MI455X (gfx1250) — 8-bit matrix operand order repaired and checked by reading, compiled, not run on hardware
//
#include <hip/hip_runtime.h>

#define BNN_EPS 1e-5f

typedef __attribute__((ext_vector_type(8))) int v8i;

__device__ __forceinline__ int permpos(int k) {
  return (((k >> 3) & 1) << 5) + ((((k >> 4) << 1) + ((k >> 2) & 1)) << 2) + (k & 3);
}

__device__ __forceinline__ v8i load_a_frag(const signed char* row, int half) {
  const int* p = (const int*)(row + half * 32);
  v8i a;
  a[0] = p[0]; a[1] = p[1]; a[2] = p[2]; a[3] = p[3];
  a[4] = p[4]; a[5] = p[5]; a[6] = p[6]; a[7] = p[7];
  return a;
}

__device__ __forceinline__ v8i load_b_frag(const signed char* nrow, int half) {
  const int* p0 = (const int*)(nrow + half * 8);
  v8i b;
  b[0] = p0[0]; b[1] = p0[1]; b[2] = p0[4]; b[3] = p0[5];
  b[4] = p0[8]; b[5] = p0[9]; b[6] = p0[12]; b[7] = p0[13];
  return b;
}

__device__ __forceinline__ v8i wmma_iu8(v8i a, v8i b, v8i c) {
  return __builtin_amdgcn_wmma_i32_16x16x64_iu8(true, a, true, b, c, false, false);
}


__global__ void k_pack_w2(const float* __restrict__ w, signed char* __restrict__ wp) {
  int i = blockIdx.x * 256 + threadIdx.x;
  if (i >= 9 * 64 * 64) return;
  int t = i >> 12, n = (i >> 6) & 63, ic = i & 63;
  wp[i] = (w[((long)n * 64 + ic) * 9 + t] >= 0.f) ? (signed char)1 : (signed char)-1;
}

__global__ void k_pack_w1(const float* __restrict__ w, signed char* __restrict__ wp) {
  long i = (long)blockIdx.x * 256 + threadIdx.x;
  if (i >= 49L * 2048 * 64) return;
  int s = (int)(i / (2048 * 64));
  int n = (int)((i >> 6) & 2047);
  int kk = (int)(i & 63);
  wp[i] = (w[(long)n * 3136 + s * 64 + kk] >= 0.f) ? (signed char)1 : (signed char)-1;
}

__global__ void k_pack_wf2(const float* __restrict__ w, signed char* __restrict__ wp) {
  int i = blockIdx.x * 256 + threadIdx.x;
  if (i >= 32 * 16 * 64) return;
  int s = i >> 10, n = (i >> 6) & 15, kk = i & 63;
  signed char v = 0;
  if (n < 10) v = (w[(long)n * 2048 + s * 64 + kk] >= 0.f) ? (signed char)1 : (signed char)-1;
  wp[i] = v;
}

__global__ void k_conv1(const float* __restrict__ x, const float* __restrict__ w1,
                        const float* __restrict__ g, const float* __restrict__ be,
                        const float* __restrict__ mu, const float* __restrict__ va,
                        signed char* __restrict__ act1) {
  int tid = threadIdx.x;
  int c = tid & 63;
  long pix = (long)blockIdx.x * 4 + (tid >> 6);
  if (pix >= 1024L * 196) return;
  int b = (int)(pix / 196), rem = (int)(pix % 196);
  int py = rem / 14, px = rem % 14;

  const float* xb = x + (long)b * 784;
  float patch[4][4];
  #pragma unroll
  for (int i = 0; i < 4; i++) {
    int iy = min(max(2 * py - 1 + i, 0), 27);
    #pragma unroll
    for (int j = 0; j < 4; j++) {
      int ix = min(max(2 * px - 1 + j, 0), 27);
      patch[i][j] = xb[iy * 28 + ix];
    }
  }
  const float* wc = w1 + c * 9;
  float s[9];
  #pragma unroll
  for (int t = 0; t < 9; t++) s[t] = (wc[t] >= 0.f) ? 1.f : -1.f;

  float sc = g[c] * rsqrtf(va[c] + BNN_EPS);
  float tc = be[c] - mu[c] * sc;

  bool pos = false;
  #pragma unroll
  for (int dy = 0; dy < 2; dy++)
    #pragma unroll
    for (int dx = 0; dx < 2; dx++) {
      float acc = 0.f;
      #pragma unroll
      for (int ky = 0; ky < 3; ky++)
        #pragma unroll
        for (int kx = 0; kx < 3; kx++)
          acc += s[ky * 3 + kx] * patch[dy + ky][dx + kx];
      if (acc * sc + tc >= 0.f) pos = true;
    }
  act1[pix * 64 + permpos(c)] = pos ? (signed char)1 : (signed char)-1;
}

__global__ void k_conv2(const signed char* __restrict__ act1,
                        const signed char* __restrict__ w2p,
                        const float* __restrict__ g, const float* __restrict__ be,
                        const float* __restrict__ mu, const float* __restrict__ va,
                        signed char* __restrict__ sgn) {
  int lane = threadIdx.x & 31;
  int wave = threadIdx.x >> 5;
  int half = lane >> 4;
  int mloc = lane & 15;
  long m0 = (long)blockIdx.x * 128 + wave * 16;
  long pix = m0 + mloc;
  int b = (int)(pix / 196), rem = (int)(pix % 196);
  int oy = rem / 14, ox = rem % 14;

  v8i acc0 = {0}, acc1 = {0}, acc2 = {0}, acc3 = {0};
  #pragma unroll
  for (int t = 0; t < 9; t++) {
    int ky = t / 3, kx = t % 3;
    int iy = min(max(oy + ky - 1, 0), 13);
    int ix = min(max(ox + kx - 1, 0), 13);
    const signed char* arow = act1 + (((long)b * 196) + iy * 14 + ix) * 64;
    v8i A = load_a_frag(arow, half);
    const signed char* bb = w2p + (long)t * 4096;
    acc0 = wmma_iu8(A, load_b_frag(bb + ( 0 + mloc) * 64, half), acc0);
    acc1 = wmma_iu8(A, load_b_frag(bb + (16 + mloc) * 64, half), acc1);
    acc2 = wmma_iu8(A, load_b_frag(bb + (32 + mloc) * 64, half), acc2);
    acc3 = wmma_iu8(A, load_b_frag(bb + (48 + mloc) * 64, half), acc3);
  }

  #pragma unroll
  for (int j = 0; j < 4; j++) {
    v8i d = (j == 0) ? acc0 : (j == 1) ? acc1 : (j == 2) ? acc2 : acc3;
    int n = j * 16 + mloc;
    float sc = g[n] * rsqrtf(va[n] + BNN_EPS);
    float tc = be[n] - mu[n] * sc;
    #pragma unroll
    for (int r = 0; r < 8; r++) {
      long p = m0 + r + 8 * half;
      float h = (float)d[r];
      sgn[p * 64 + n] = (h * sc + tc >= 0.f) ? (signed char)1 : (signed char)0;
    }
  }
}

__global__ void k_pool2(const signed char* __restrict__ sgn, signed char* __restrict__ act2) {
  long i = (long)blockIdx.x * 256 + threadIdx.x;
  if (i >= 1024L * 64 * 49) return;
  int b = (int)(i / (64 * 49));
  int r = (int)(i % (64 * 49));
  int c = r / 49, q = r % 49;
  int y2 = q / 7, x2 = q % 7;
  const signed char* base = sgn + (long)b * 196 * 64;
  int v = 0;
  #pragma unroll
  for (int dy = 0; dy < 2; dy++)
    #pragma unroll
    for (int dx = 0; dx < 2; dx++)
      v |= base[((2 * y2 + dy) * 14 + (2 * x2 + dx)) * 64 + c];
  int k = c * 49 + q;
  act2[(long)b * 3136 + (k & ~63) + permpos(k & 63)] =
      v ? (signed char)1 : (signed char)-1;
}

__global__ void k_fc1(const signed char* __restrict__ act2,
                      const signed char* __restrict__ w1p,
                      const float* __restrict__ g, const float* __restrict__ be,
                      const float* __restrict__ mu, const float* __restrict__ va,
                      signed char* __restrict__ act3) {
  int lane = threadIdx.x & 31;
  int wave = threadIdx.x >> 5;
  int half = lane >> 4;
  int mloc = lane & 15;
  long m0 = (long)blockIdx.x * 128 + wave * 16;
  int n0 = blockIdx.y * 64;

  const signed char* arow = act2 + (m0 + mloc) * 3136;
  v8i acc0 = {0}, acc1 = {0}, acc2 = {0}, acc3 = {0};
  for (int s = 0; s < 49; s++) {
    v8i A = load_a_frag(arow + s * 64, half);
    const signed char* bb = w1p + ((long)s * 2048 + n0) * 64;
    acc0 = wmma_iu8(A, load_b_frag(bb + ( 0 + mloc) * 64, half), acc0);
    acc1 = wmma_iu8(A, load_b_frag(bb + (16 + mloc) * 64, half), acc1);
    acc2 = wmma_iu8(A, load_b_frag(bb + (32 + mloc) * 64, half), acc2);
    acc3 = wmma_iu8(A, load_b_frag(bb + (48 + mloc) * 64, half), acc3);
  }

  #pragma unroll
  for (int j = 0; j < 4; j++) {
    v8i d = (j == 0) ? acc0 : (j == 1) ? acc1 : (j == 2) ? acc2 : acc3;
    int n = n0 + j * 16 + mloc;
    float sc = g[n] * rsqrtf(va[n] + BNN_EPS);
    float tc = be[n] - mu[n] * sc;
    int inner = permpos(n & 63);
    #pragma unroll
    for (int r = 0; r < 8; r++) {
      long bi = m0 + r + 8 * half;
      float h = (float)d[r];
      act3[bi * 2048 + (n & ~63) + inner] = (h * sc + tc >= 0.f) ? (signed char)1 : (signed char)-1;
    }
  }
}

__global__ void k_fc2(const signed char* __restrict__ act3,
                      const signed char* __restrict__ wfp,
                      const float* __restrict__ scale, float* __restrict__ out) {
  int lane = threadIdx.x & 31;
  int wave = threadIdx.x >> 5;
  int half = lane >> 4;
  int mloc = lane & 15;
  long m0 = (long)blockIdx.x * 128 + wave * 16;

  const signed char* arow = act3 + (m0 + mloc) * 2048;
  v8i acc = {0};
  for (int s = 0; s < 32; s++) {
    v8i A = load_a_frag(arow + s * 64, half);
    v8i B = load_b_frag(wfp + (s * 16 + mloc) * 64, half);
    acc = wmma_iu8(A, B, acc);
  }
  float sv = scale[0];
  int o = mloc;
  if (o < 10) {
    #pragma unroll
    for (int r = 0; r < 8; r++) {
      long bi = m0 + r + 8 * half;
      out[bi * 10 + o] = (float)acc[r] * sv;
    }
  }
}

extern "C" void kernel_launch(void* const* d_in, const int* in_sizes, int n_in,
                              void* d_out, int out_size, void* d_ws, size_t ws_size,
                              hipStream_t stream) {
  const float* x    = (const float*)d_in[0];
  const float* w1   = (const float*)d_in[1];
  const float* g1   = (const float*)d_in[2];
  const float* b1   = (const float*)d_in[3];
  const float* m1   = (const float*)d_in[4];
  const float* v1   = (const float*)d_in[5];
  const float* w2   = (const float*)d_in[6];
  const float* g2   = (const float*)d_in[7];
  const float* b2   = (const float*)d_in[8];
  const float* m2   = (const float*)d_in[9];
  const float* v2   = (const float*)d_in[10];
  const float* fw1  = (const float*)d_in[11];
  const float* g3   = (const float*)d_in[12];
  const float* b3   = (const float*)d_in[13];
  const float* m3   = (const float*)d_in[14];
  const float* v3   = (const float*)d_in[15];
  const float* fw2  = (const float*)d_in[16];
  const float* scal = (const float*)d_in[17];
  float* out = (float*)d_out;

  char* ws = (char*)d_ws;
  signed char* act1 = (signed char*)(ws);
  signed char* sgn  = act1 + 12845056;
  signed char* act2 = sgn  + 12845056;
  signed char* act3 = act2 + 3211264;
  signed char* w2p  = act3 + 2097152;
  signed char* w1p  = w2p  + 36864;
  signed char* wfp  = w1p  + 6422528;

  k_pack_w2 <<<(9 * 64 * 64 + 255) / 256, 256, 0, stream>>>(w2, w2p);
  k_pack_w1 <<<(int)((49L * 2048 * 64 + 255) / 256), 256, 0, stream>>>(fw1, w1p);
  k_pack_wf2<<<(32 * 16 * 64 + 255) / 256, 256, 0, stream>>>(fw2, wfp);

  k_conv1<<<50176, 256, 0, stream>>>(x, w1, g1, b1, m1, v1, act1);
  k_conv2<<<1568, 256, 0, stream>>>(act1, w2p, g2, b2, m2, v2, sgn);
  k_pool2<<<12544, 256, 0, stream>>>(sgn, act2);
  k_fc1<<<dim3(8, 32), 256, 0, stream>>>(act2, w1p, g3, b3, m3, v3, act3);
  k_fc2<<<8, 256, 0, stream>>>(act3, wfp, scal, out);
}
